// FusionNeRF_28638841929924
// MI455X (gfx1250) — compile-verified, matrix operand order checked by reading, not run on hardware
//
#include <hip/hip_runtime.h>

typedef __attribute__((ext_vector_type(16))) _Float16 v16h;
typedef __attribute__((ext_vector_type(8)))  float    v8f;
typedef __attribute__((ext_vector_type(4)))  int      v4i;

#define HD 256
#define SS 64
#define NRAYS 4096
#define NBLK  2048            // 2 rays per block, M = 128 rows
#define MM    128

// d_ws layout (half elements): transposed+padded f16 weights
#define OFF_SWIN   0                 // [256 n][32 k]
#define OFF_DWIN   8192              // [256 n][32 k]
#define OFF_SWH    16384             // [3][256 n][256 k]
#define OFF_DWH    212992            // [3][256 n][256 k]
#define OFF_SWOUT  409600            // [16 n][256 k]
#define OFF_DWOUT  413696            // [16 n][256 k]
#define WS_HALVES  417792            // 835,584 bytes

// d_out offsets (floats), outputs concatenated in return order
#define OUT_RGB    0                 // [4096,3]
#define OUT_DEPTH  12288             // [4096]
#define OUT_W      16384             // [4096,64]
#define OUT_SW     278528            // [4096,64]
#define OUT_DW     540672            // [4096,64]

// LDS: 2*(128*256*2) + 256*256*2 + 2*(128*16*4) + 768*4 = 281600 B (< 320KB/WGP)
#define SMEM_BYTES 281600

#if defined(__HIP_DEVICE_COMPILE__) && \
    __has_builtin(__builtin_amdgcn_global_load_async_to_lds_b128) && \
    __has_builtin(__builtin_amdgcn_s_wait_asynccnt)
#define USE_ASYNC_LDS 1
#else
#define USE_ASYNC_LDS 0
#endif

union Frag16 { v16h v; uint4 q[2]; };

__device__ __forceinline__ float sigm(float x) { return 1.0f / (1.0f + expf(-x)); }

// ---------------- weight conversion: f32 -> transposed, padded f16 ----------------
__global__ void wconvert_kernel(const float* __restrict__ sW_in, const float* __restrict__ sW_h,
                                const float* __restrict__ sW_out, const float* __restrict__ dW_in,
                                const float* __restrict__ dW_h, const float* __restrict__ dW_out,
                                _Float16* __restrict__ ws) {
  int i = blockIdx.x * blockDim.x + threadIdx.x;
  if (i >= WS_HALVES) return;
  float v = 0.0f;
  if (i < OFF_DWIN) {                         // static input layer, pad K 6->32
    int n = i >> 5, k = i & 31;
    if (k < 6) v = sW_in[k * HD + n];
  } else if (i < OFF_SWH) {                   // dynamic input layer, pad K 7->32
    int j = i - OFF_DWIN; int n = j >> 5, k = j & 31;
    if (k < 7) v = dW_in[k * HD + n];
  } else if (i < OFF_DWH) {                   // static hidden [l][n][k] <- [l][k][n]
    int j = i - OFF_SWH; int l = j >> 16, r = j & 65535, n = r >> 8, k = r & 255;
    v = sW_h[l * 65536 + k * HD + n];
  } else if (i < OFF_SWOUT) {                 // dynamic hidden
    int j = i - OFF_DWH; int l = j >> 16, r = j & 65535, n = r >> 8, k = r & 255;
    v = dW_h[l * 65536 + k * HD + n];
  } else if (i < OFF_DWOUT) {                 // static out, pad N 4->16
    int j = i - OFF_SWOUT; int n = j >> 8, k = j & 255;
    if (n < 4) v = sW_out[k * 4 + n];
  } else {                                    // dynamic out, pad N 5->16
    int j = i - OFF_DWOUT; int n = j >> 8, k = j & 255;
    if (n < 5) v = dW_out[k * 5 + n];
  }
  ws[i] = (_Float16)v;
}

// ---------------- LDS staging (async if available) + prefetch ----------------
__device__ __forceinline__ void stage_w(_Float16* dst, const _Float16* __restrict__ src,
                                        int halves, int tid) {
  const int chunks = halves >> 3;               // 16B chunks
#if USE_ASYNC_LDS
  for (int c = tid; c < chunks; c += 256) {
    __builtin_amdgcn_global_load_async_to_lds_b128(
        (__attribute__((address_space(1))) v4i*)(src + (c << 3)),
        (__attribute__((address_space(3))) v4i*)(dst + (c << 3)), 0, 0);
  }
#else
  for (int c = tid; c < chunks; c += 256)
    *(uint4*)(dst + (c << 3)) = *(const uint4*)(src + (c << 3));
#endif
}

__device__ __forceinline__ void stage_wait() {
#if USE_ASYNC_LDS
  __builtin_amdgcn_s_wait_asynccnt(0);
#endif
}

__device__ __forceinline__ void prefetch_w(const _Float16* src, int halves, int tid) {
  for (int c = tid * 64; c < halves; c += 256 * 64)   // one 128B line per thread per step
    __builtin_prefetch(src + c, 0, 1);                 // -> global_prefetch_b8
}

// ---------------- GEMM layers ----------------
// A: [128][KD] f16 acts in LDS, B: [N][KD] f16 W^T in LDS.
// 8 row-tiles == 8 waves: each wave owns one row-tile, keeps all A fragments in
// VGPRs, and streams the 16 col-tiles (B fragments) from LDS.
template<int KD>
__device__ __forceinline__ void gemm_layer(const _Float16* hIn, const _Float16* wT,
                                           const float* __restrict__ bias, _Float16* hOut,
                                           int waveId, int lane) {
  const int ml = lane & 15, kh = lane >> 4;
  constexpr int KC = KD / 32;
  Frag16 a[KC];
  const _Float16* ar = hIn + (waveId * 16 + ml) * KD + kh * 8;
#pragma unroll
  for (int kc = 0; kc < KC; ++kc) {            // load A fragments once per layer
    a[kc].q[0] = *(const uint4*)(ar + kc * 32);
    a[kc].q[1] = *(const uint4*)(ar + kc * 32 + 16);
  }
#pragma unroll 1
  for (int nt = 0; nt < 16; ++nt) {            // stream col-tiles
    v8f acc = {};
#pragma unroll
    for (int kc = 0; kc < KC; ++kc) {
      Frag16 b;
      const _Float16* bp = wT + (nt * 16 + ml) * KD + kc * 32 + kh * 8;
      b.q[0] = *(const uint4*)bp;
      b.q[1] = *(const uint4*)(bp + 16);
      acc = __builtin_amdgcn_wmma_f32_16x16x32_f16(false, a[kc].v, false, b.v,
                                                   (short)0, acc, false, false);
    }
    const int n = nt * 16 + ml;
    const float bv = bias[n];
#pragma unroll
    for (int r = 0; r < 8; ++r) {              // C/D layout: vgpr r -> M = r + 8*(lane>>4)
      float x = fmaxf(acc[r] + bv, 0.0f);
      hOut[(waveId * 16 + r + kh * 8) * HD + n] = (_Float16)x;
    }
  }
}

__device__ __forceinline__ void gemm_out(const _Float16* hIn, const _Float16* wT,
                                         const float* __restrict__ bias, int biasN,
                                         float* obuf, int waveId, int lane) {
  const int ml = lane & 15, kh = lane >> 4;    // 8 row-tiles, 1 col-tile (N=16)
  v8f acc = {};
#pragma unroll
  for (int kc = 0; kc < 8; ++kc) {
    Frag16 a, b;
    const _Float16* ap = hIn + (waveId * 16 + ml) * HD + kc * 32 + kh * 8;
    a.q[0] = *(const uint4*)ap;
    a.q[1] = *(const uint4*)(ap + 16);
    const _Float16* bp = wT + ml * HD + kc * 32 + kh * 8;
    b.q[0] = *(const uint4*)bp;
    b.q[1] = *(const uint4*)(bp + 16);
    acc = __builtin_amdgcn_wmma_f32_16x16x32_f16(false, a.v, false, b.v,
                                                 (short)0, acc, false, false);
  }
  const float bv = (ml < biasN) ? bias[ml] : 0.0f;
#pragma unroll
  for (int r = 0; r < 8; ++r)
    obuf[(waveId * 16 + r + kh * 8) * 16 + ml] = acc[r] + bv;
}

// ---------------- fused NeRF kernel: 1 block = 2 rays (128 samples) ----------------
__global__ __launch_bounds__(256, 1)
void nerf_fused_kernel(const float* __restrict__ points, const float* __restrict__ dirs,
                       const float* __restrict__ z_vals, const float* __restrict__ tim,
                       const float* __restrict__ sb_in, const float* __restrict__ sb_h,
                       const float* __restrict__ sb_out, const float* __restrict__ db_in,
                       const float* __restrict__ db_h, const float* __restrict__ db_out,
                       const _Float16* __restrict__ ws, float* __restrict__ out) {
  const int tid = threadIdx.x;
  const int lane = tid & 31;
  const int waveId = tid >> 5;
  const size_t base = (size_t)blockIdx.x * MM;   // first (ray,sample) flat index

  extern __shared__ char smem[];
  _Float16* hA = (_Float16*)smem;            // 128x256 acts (ping)   64KB
  _Float16* hB = hA + MM * HD;               // 128x256 acts (pong)   64KB
  _Float16* wS = hB + MM * HD;               // 256x256 W^T tile     128KB
  float* obS = (float*)(wS + HD * HD);       // 128x16 static out      8KB
  float* obD = obS + MM * 16;                // 128x16 dynamic out     8KB
  float* ep  = obD + MM * 16;                // epilogue scratch       3KB

  // ---------- static MLP ----------
  if (tid < MM) {                            // build x_s rows [128][32]
    const float* p = points + (base + tid) * 3;
    const float* d = dirs   + (base + tid) * 3;
    _Float16* row = hA + tid * 32;
#pragma unroll
    for (int c = 0; c < 32; ++c) row[c] = (_Float16)0.0f;
    row[0] = (_Float16)p[0]; row[1] = (_Float16)p[1]; row[2] = (_Float16)p[2];
    row[3] = (_Float16)d[0]; row[4] = (_Float16)d[1]; row[5] = (_Float16)d[2];
  }
  stage_w(wS, ws + OFF_SWIN, 8192, tid);
  prefetch_w(ws + OFF_SWH, 65536, tid);
  stage_wait();
  __syncthreads();
  gemm_layer<32>(hA, wS, sb_in, hB, waveId, lane);
  __syncthreads();
  stage_w(wS, ws + OFF_SWH, 65536, tid);
  prefetch_w(ws + OFF_SWH + 65536, 65536, tid);
  stage_wait();
  __syncthreads();
  gemm_layer<256>(hB, wS, sb_h, hA, waveId, lane);
  __syncthreads();
  stage_w(wS, ws + OFF_SWH + 65536, 65536, tid);
  prefetch_w(ws + OFF_SWH + 131072, 65536, tid);
  stage_wait();
  __syncthreads();
  gemm_layer<256>(hA, wS, sb_h + 256, hB, waveId, lane);
  __syncthreads();
  stage_w(wS, ws + OFF_SWH + 131072, 65536, tid);
  stage_wait();
  __syncthreads();
  gemm_layer<256>(hB, wS, sb_h + 512, hA, waveId, lane);
  __syncthreads();
  stage_w(wS, ws + OFF_SWOUT, 4096, tid);
  stage_wait();
  __syncthreads();
  gemm_out(hA, wS, sb_out, 4, obS, waveId, lane);
  __syncthreads();

  // ---------- dynamic MLP ----------
  if (tid < MM) {                            // build x_d rows [128][32] (adds time)
    const float* p = points + (base + tid) * 3;
    const float* d = dirs   + (base + tid) * 3;
    _Float16* row = hA + tid * 32;
#pragma unroll
    for (int c = 0; c < 32; ++c) row[c] = (_Float16)0.0f;
    row[0] = (_Float16)p[0]; row[1] = (_Float16)p[1]; row[2] = (_Float16)p[2];
    row[3] = (_Float16)d[0]; row[4] = (_Float16)d[1]; row[5] = (_Float16)d[2];
    row[6] = (_Float16)tim[base + tid];
  }
  stage_w(wS, ws + OFF_DWIN, 8192, tid);
  prefetch_w(ws + OFF_DWH, 65536, tid);
  stage_wait();
  __syncthreads();
  gemm_layer<32>(hA, wS, db_in, hB, waveId, lane);
  __syncthreads();
  stage_w(wS, ws + OFF_DWH, 65536, tid);
  prefetch_w(ws + OFF_DWH + 65536, 65536, tid);
  stage_wait();
  __syncthreads();
  gemm_layer<256>(hB, wS, db_h, hA, waveId, lane);
  __syncthreads();
  stage_w(wS, ws + OFF_DWH + 65536, 65536, tid);
  prefetch_w(ws + OFF_DWH + 131072, 65536, tid);
  stage_wait();
  __syncthreads();
  gemm_layer<256>(hA, wS, db_h + 256, hB, waveId, lane);
  __syncthreads();
  stage_w(wS, ws + OFF_DWH + 131072, 65536, tid);
  stage_wait();
  __syncthreads();
  gemm_layer<256>(hB, wS, db_h + 512, hA, waveId, lane);
  __syncthreads();
  stage_w(wS, ws + OFF_DWOUT, 4096, tid);
  stage_wait();
  __syncthreads();
  gemm_out(hA, wS, db_out, 5, obD, waveId, lane);
  __syncthreads();

  // ---------- epilogue: blend, alpha, per-ray transmittance scan, reductions ----------
  float* zv = ep;             // [128] z values
  float* tb = ep + 128;       // [128] scan buffer (1 - alpha + eps)
  float* red = ep + 256;      // [4][128] reduction arrays
  float alpha = 0.f, w = 0.f, bw = 0.f, rr = 0.f, rg = 0.f, rb = 0.f, z = 0.f;
  const int s = tid & 63;     // sample within ray

  if (tid < MM) { z = z_vals[base + tid]; zv[tid] = z; }
  __syncthreads();
  if (tid < MM) {
    const float ssig = obS[tid * 16 + 0];
    const float sr0 = sigm(obS[tid * 16 + 1]), sr1 = sigm(obS[tid * 16 + 2]),
                sr2 = sigm(obS[tid * 16 + 3]);
    const float dsig = obD[tid * 16 + 0];
    const float dr0 = sigm(obD[tid * 16 + 1]), dr1 = sigm(obD[tid * 16 + 2]),
                dr2 = sigm(obD[tid * 16 + 3]);
    bw = sigm(obD[tid * 16 + 4]);
    const float sigma = (1.f - bw) * ssig + bw * dsig;
    rr = (1.f - bw) * sr0 + bw * dr0;
    rg = (1.f - bw) * sr1 + bw * dr1;
    rb = (1.f - bw) * sr2 + bw * dr2;
    const float delta = (s < SS - 1) ? (zv[tid + 1] - zv[tid]) : 1e10f;
    alpha = 1.f - expf(-sigma * delta);
    tb[tid] = 1.f - alpha + 1e-10f;
  }
  __syncthreads();
  // Hillis-Steele inclusive prefix product, segmented per ray (64-sample segments)
  for (int off = 1; off < SS; off <<= 1) {
    float vprev = 0.f;
    const bool act = (tid < MM) && (s >= off);
    if (act) vprev = tb[tid - off];
    __syncthreads();
    if (act) tb[tid] *= vprev;
    __syncthreads();
  }
  if (tid < MM) {
    const float trans = (s == 0) ? 1.f : tb[tid - 1];
    w = alpha * trans;
    out[OUT_W  + base + tid] = w;
    out[OUT_SW + base + tid] = (1.f - bw) * w;
    out[OUT_DW + base + tid] = bw * w;
    red[tid]        = w * rr;
    red[128 + tid]  = w * rg;
    red[256 + tid]  = w * rb;
    red[384 + tid]  = w * z;
  }
  __syncthreads();
  if (tid < 8) {                               // 2 rays x {r,g,b,depth}
    const int rl = tid >> 2, c = tid & 3;
    const float* a = red + c * 128 + rl * 64;
    float acc = 0.f;
    for (int i = 0; i < SS; ++i) acc += a[i];
    const int ray = blockIdx.x * 2 + rl;
    if (c < 3) out[OUT_RGB + (size_t)ray * 3 + c] = acc;
    else       out[OUT_DEPTH + ray] = acc;
  }
}

extern "C" void kernel_launch(void* const* d_in, const int* in_sizes, int n_in,
                              void* d_out, int out_size, void* d_ws, size_t ws_size,
                              hipStream_t stream) {
  (void)in_sizes; (void)n_in; (void)out_size; (void)ws_size;
  const float* points = (const float*)d_in[0];
  const float* dirs   = (const float*)d_in[1];
  const float* z_vals = (const float*)d_in[2];
  const float* tim    = (const float*)d_in[3];
  const float* sW_in  = (const float*)d_in[4];
  const float* sb_in  = (const float*)d_in[5];
  const float* sW_h   = (const float*)d_in[6];
  const float* sb_h   = (const float*)d_in[7];
  const float* sW_out = (const float*)d_in[8];
  const float* sb_out = (const float*)d_in[9];
  const float* dW_in  = (const float*)d_in[10];
  const float* db_in  = (const float*)d_in[11];
  const float* dW_h   = (const float*)d_in[12];
  const float* db_h   = (const float*)d_in[13];
  const float* dW_out = (const float*)d_in[14];
  const float* db_out = (const float*)d_in[15];
  _Float16* ws = (_Float16*)d_ws;
  float* out = (float*)d_out;

  (void)hipFuncSetAttribute(reinterpret_cast<const void*>(nerf_fused_kernel),
                            hipFuncAttributeMaxDynamicSharedMemorySize, SMEM_BYTES);

  wconvert_kernel<<<(WS_HALVES + 255) / 256, 256, 0, stream>>>(
      sW_in, sW_h, sW_out, dW_in, dW_h, dW_out, ws);
  nerf_fused_kernel<<<NBLK, 256, SMEM_BYTES, stream>>>(
      points, dirs, z_vals, tim,
      sb_in, sb_h, sb_out, db_in, db_h, db_out, ws, out);
}
